// RGCN_3229815407101
// MI455X (gfx1250) — matrix operand order repaired and checked by reading, not compiled, not run on hardware
//
#include <hip/hip_runtime.h>

#define DD      256
#define NNODES  20000
#define NEDGES  320000
#define RREL    8
#define NMATS   9              // 8 relations + self
#define MTILES  (NNODES / 16)  // 1250

typedef __attribute__((ext_vector_type(16))) __bf16         v16bf;
typedef __attribute__((ext_vector_type(8)))  float          v8f;
typedef __attribute__((ext_vector_type(8)))  unsigned short v8u16;
typedef __attribute__((ext_vector_type(16))) unsigned short v16u16;

__device__ __forceinline__ unsigned short f32_to_bf16(float f) {
    unsigned int u = __float_as_uint(f);
    u += 0x7FFFu + ((u >> 16) & 1u);   // round-to-nearest-even
    return (unsigned short)(u >> 16);
}

// ---------------------------------------------------------------------------
// Convert all 18 weight matrices (2 layers x [8 relations + self]) to bf16,
// stored TRANSPOSED: wt[mat][o][i] = W[mat][i][o].  This makes the WMMA
// B-fragment (per-lane column n, contiguous K) a contiguous 32B load.
// ---------------------------------------------------------------------------
__global__ void rgcn_wconv(const float* __restrict__ W1, const float* __restrict__ S1,
                           const float* __restrict__ W2, const float* __restrict__ S2,
                           unsigned short* __restrict__ wt) {
    int idx = blockIdx.x * blockDim.x + threadIdx.x;    // 18 * 65536 threads
    int mm  = idx >> 16;
    int rem = idx & 65535;
    int o   = rem >> 8;     // output column
    int i   = rem & 255;    // K index
    int layer = mm / NMATS;
    int r     = mm % NMATS;
    const float* Wl = layer ? W2 : W1;
    const float* Sl = layer ? S2 : S1;
    const float* srcp = (r < RREL) ? (Wl + (size_t)r * DD * DD) : Sl;
    wt[(size_t)mm * DD * DD + (size_t)o * DD + i] = f32_to_bf16(srcp[(size_t)i * DD + o]);
}

// f32 -> bf16 cast (optionally fused ReLU)
__global__ void rgcn_cast(const float* __restrict__ in, unsigned short* __restrict__ out,
                          int n, int relu) {
    int i = blockIdx.x * 256 + threadIdx.x;
    if (i < n) {
        float v = in[i];
        if (relu) v = fmaxf(v, 0.0f);
        out[i] = f32_to_bf16(v);
    }
}

// ---------------------------------------------------------------------------
// 9-way batched GEMM:  out[mat] = Xbf16 @ W[mat]   (f32 accumulate via WMMA)
//   mats 0..7  -> msg[r][n][d]            (relation messages)
//   mat  8     -> agg[n][d] = self + bias (also initializes the accumulator)
// One wave computes a 16x64 output strip: 4 x v_wmma_f32_16x16x32_bf16 per
// K-step of 32, A fragment reused across the 4 N-tiles.
// ---------------------------------------------------------------------------
__global__ __launch_bounds__(256) void rgcn_gemm9(
    const unsigned short* __restrict__ xb,   // [NNODES][DD] bf16
    const unsigned short* __restrict__ wt,   // [NMATS][DD][DD] bf16, transposed
    const float*          __restrict__ bias, // [DD]
    float*                __restrict__ msg,  // [RREL][NNODES][DD]
    float*                __restrict__ agg)  // [NNODES][DD]
{
    const int gwave = (int)((blockIdx.x * 256u + threadIdx.x) >> 5);
    const int lane  = threadIdx.x & 31;

    const int nstrip = gwave & 3;            // 4 strips of 64 cols
    int t            = gwave >> 2;
    const int mtile  = t % MTILES;
    const int mat    = t / MTILES;           // grid sized exactly: mat in [0,9)

    const int m0 = mtile * 16;
    const int n0 = nstrip * 64;

    // A fragment addressing (16-bit A 16x32, contiguous K order used here, not ISA table order):
    //   lane L: row m = L&15; lane group selects K base 0 or 16;
    //   halves 0..15 = K=ak..ak+15 (contiguous, matches B)
    const int am = lane & 15;
    const int ak = (lane >> 4) << 4;
    const unsigned short* arow = xb + (size_t)(m0 + am) * DD;

    // B fragment addressing: lane L: col n = L&15; K base 0 or 16; 16 contiguous K
    const int bn = lane & 15;
    const int bk = (lane >> 4) << 4;
    const unsigned short* bbase =
        wt + (size_t)mat * DD * DD + (size_t)(n0 + bn) * DD + bk;

    v8f acc0 = {}, acc1 = {}, acc2 = {}, acc3 = {};

    #pragma unroll
    for (int k = 0; k < DD; k += 32) {
        v8u16 alo = *(const v8u16*)(arow + k + ak);
        v8u16 ahi = *(const v8u16*)(arow + k + ak + 8);
        v16u16 au = __builtin_shufflevector(alo, ahi,
                        0,1,2,3,4,5,6,7,8,9,10,11,12,13,14,15);
        v16bf a = __builtin_bit_cast(v16bf, au);

        v16bf b0 = __builtin_bit_cast(v16bf, *(const v16u16*)(bbase + (size_t)0 * 16 * DD + k));
        v16bf b1 = __builtin_bit_cast(v16bf, *(const v16u16*)(bbase + (size_t)1 * 16 * DD + k));
        v16bf b2 = __builtin_bit_cast(v16bf, *(const v16u16*)(bbase + (size_t)2 * 16 * DD + k));
        v16bf b3 = __builtin_bit_cast(v16bf, *(const v16u16*)(bbase + (size_t)3 * 16 * DD + k));

        acc0 = __builtin_amdgcn_wmma_f32_16x16x32_bf16(false, a, false, b0, (short)0, acc0, false, false);
        acc1 = __builtin_amdgcn_wmma_f32_16x16x32_bf16(false, a, false, b1, (short)0, acc1, false, false);
        acc2 = __builtin_amdgcn_wmma_f32_16x16x32_bf16(false, a, false, b2, (short)0, acc2, false, false);
        acc3 = __builtin_amdgcn_wmma_f32_16x16x32_bf16(false, a, false, b3, (short)0, acc3, false, false);
    }

    // C/D layout: lane L: col n = L&15; rows (L>>4)*8 + v for VGPR v
    const int cn = lane & 15;
    const int cm = (lane >> 4) << 3;

    if (mat < RREL) {
        float* ob = msg + ((size_t)mat * NNODES + m0 + cm) * DD + n0 + cn;
        #pragma unroll
        for (int v = 0; v < 8; ++v) {
            ob[(size_t)v * DD +  0] = acc0[v];
            ob[(size_t)v * DD + 16] = acc1[v];
            ob[(size_t)v * DD + 32] = acc2[v];
            ob[(size_t)v * DD + 48] = acc3[v];
        }
    } else {
        float bb0 = bias[n0 +  0 + cn];
        float bb1 = bias[n0 + 16 + cn];
        float bb2 = bias[n0 + 32 + cn];
        float bb3 = bias[n0 + 48 + cn];
        float* ob = agg + (size_t)(m0 + cm) * DD + n0 + cn;
        #pragma unroll
        for (int v = 0; v < 8; ++v) {
            ob[(size_t)v * DD +  0] = acc0[v] + bb0;
            ob[(size_t)v * DD + 16] = acc1[v] + bb1;
            ob[(size_t)v * DD + 32] = acc2[v] + bb2;
            ob[(size_t)v * DD + 48] = acc3[v] + bb3;
        }
    }
}

// ---------------------------------------------------------------------------
// Edge scatter: agg[dst] += msg[etype][src].  64 threads/edge, float4 gather
// (coalesced 128b loads) + 4 x global_atomic_add_f32.
// ---------------------------------------------------------------------------
__global__ __launch_bounds__(256) void rgcn_scatter(
    const float* __restrict__ msg,
    const int*   __restrict__ src,
    const int*   __restrict__ dst,
    const int*   __restrict__ ety,
    float*       __restrict__ agg)
{
    int e = blockIdx.x * 4 + (threadIdx.x >> 6);   // 4 edges per block
    int t = threadIdx.x & 63;                      // 64 threads x float4 = 256 floats
    int s = src[e];
    int d = dst[e];
    int r = ety[e];
    const float4 v = *(const float4*)(msg + ((size_t)r * NNODES + s) * DD + t * 4);
    float* p = agg + (size_t)d * DD + t * 4;
    atomicAdd(p + 0, v.x);
    atomicAdd(p + 1, v.y);
    atomicAdd(p + 2, v.z);
    atomicAdd(p + 3, v.w);
}

// ---------------------------------------------------------------------------
extern "C" void kernel_launch(void* const* d_in, const int* in_sizes, int n_in,
                              void* d_out, int out_size, void* d_ws, size_t ws_size,
                              hipStream_t stream) {
    const float* x   = (const float*)d_in[0];
    const float* W1  = (const float*)d_in[1];
    const float* S1  = (const float*)d_in[2];
    const float* b1  = (const float*)d_in[3];
    const float* W2  = (const float*)d_in[4];
    const float* S2  = (const float*)d_in[5];
    const float* b2  = (const float*)d_in[6];
    const int*   eix = (const int*)d_in[7];      // [2][NEDGES]
    const int*   ety = (const int*)d_in[8];
    const int*   src = eix;
    const int*   dst = eix + NEDGES;
    float*       out = (float*)d_out;

    // Workspace layout (~197 MB):
    char* ws = (char*)d_ws;
    float* msg = (float*)ws;                                          // R*N*D f32
    char*  p   = ws + (size_t)RREL * NNODES * DD * sizeof(float);
    float* agg1 = (float*)p;                                          // N*D f32
    p += (size_t)NNODES * DD * sizeof(float);
    unsigned short* hb = (unsigned short*)p;                          // N*D bf16
    p += (size_t)NNODES * DD * sizeof(unsigned short);
    unsigned short* wt = (unsigned short*)p;                          // 18*D*D bf16

    const int gemmBlocks = (NMATS * MTILES * 4) / 8;   // 45000 waves / 8 per block = 5625
    const int castBlocks = (NNODES * DD) / 256;        // 20000
    const int scatBlocks = NEDGES / 4;                 // 80000

    // Weight prep (both layers, bf16 + transpose)
    rgcn_wconv<<<(2 * NMATS * DD * DD) / 256, 256, 0, stream>>>(W1, S1, W2, S2, wt);

    // ---- Layer 1 ----
    rgcn_cast<<<castBlocks, 256, 0, stream>>>(x, hb, NNODES * DD, 0);
    rgcn_gemm9<<<gemmBlocks, 256, 0, stream>>>(hb, wt, b1, msg, agg1);
    rgcn_scatter<<<scatBlocks, 256, 0, stream>>>(msg, src, dst, ety, agg1);

    // ---- ReLU + Layer 2 ----
    rgcn_cast<<<castBlocks, 256, 0, stream>>>(agg1, hb, NNODES * DD, 1);
    rgcn_gemm9<<<gemmBlocks, 256, 0, stream>>>(hb, wt + (size_t)NMATS * DD * DD, b2, msg, out);
    rgcn_scatter<<<scatBlocks, 256, 0, stream>>>(msg, src, dst, ety, out);
}
